// EleutherEncoderDecoder_43130061586702
// MI455X (gfx1250) — matrix operand order repaired and checked by reading, not compiled, not run on hardware
//
#include <hip/hip_runtime.h>
#include <stdint.h>

#define B_TOK   4096
#define D_IN    2048
#define D_SAE   32768
#define K_TOP   64

#define M_TILE  32      // token rows per workgroup
#define N_CHUNK 128     // latents per iteration (8 waves x 16)

typedef __attribute__((ext_vector_type(16))) __bf16 v16bf;
typedef __attribute__((ext_vector_type(8)))  float  v8f;

// ---- LDS layout (bytes) ----
// xtile: [64 k-chunks][32 rows][64B]  (bf16, 32 K-values per row-chunk, XOR-swizzled)
#define XTILE_BYTES   (64 * M_TILE * 64)          // 131072
#define STAGE_OFF     XTILE_BYTES
#define STAGE_BYTES   (M_TILE * 129 * 4)          // 16512  (pitch 129 -> conflict-free col scan)
#define TOPV_OFF      (STAGE_OFF + STAGE_BYTES)
#define TOPV_BYTES    (M_TILE * 65 * 4)           // 8320
#define TOPI_OFF      (TOPV_OFF + TOPV_BYTES)
#define TOPI_BYTES    (M_TILE * 65 * 2)           // 4160
#define LDS_TOTAL     (TOPI_OFF + TOPI_BYTES)     // 160064  -> 2 WGs / WGP

// ============================================================================
// Kernel 1: fused encoder GEMM (bf16 WMMA, f32 accum) + per-row top-64
// ============================================================================
__global__ __launch_bounds__(256)
void sae_encode_topk(const float* __restrict__ x,
                     const float* __restrict__ W_enc,
                     const float* __restrict__ b_enc,
                     const float* __restrict__ b_dec,
                     float* __restrict__ out_vals,
                     int*   __restrict__ out_idx)
{
    extern __shared__ char smem[];
    char*           xtile = smem;
    float*          stage = (float*)(smem + STAGE_OFF);
    float*          topv  = (float*)(smem + TOPV_OFF);
    unsigned short* topi  = (unsigned short*)(smem + TOPI_OFF);

    const int tid  = threadIdx.x;
    const int lane = tid & 31;
    const int wave = tid >> 5;        // 0..7
    const int l16  = lane & 15;
    const int hi16 = lane >> 4;       // 0 or 1
    const int row0 = blockIdx.x * M_TILE;

    // ---- Stage xc = bf16(x - b_dec) into LDS, pre-swizzled for WMMA A-frags.
    // Logical layout per (kc,row): 64B = K values [0..31] as bf16. Physical 16B
    // sub-chunks XORed by ((row&3)<<4) to spread LDS banks.
    for (int c = tid; c < M_TILE * 64; c += 256) {
        const int r  = c >> 6;        // consecutive tids -> consecutive kc: coalesced
        const int kc = c & 63;
        const float* xs = x + (size_t)(row0 + r) * D_IN + kc * 32;
        const float* bs = b_dec + kc * 32;
        __bf16 tmp[32];
#pragma unroll
        for (int i = 0; i < 32; ++i) tmp[i] = (__bf16)(xs[i] - bs[i]);
        const uint4* tq = (const uint4*)tmp;
        char* dst = xtile + ((kc * M_TILE + r) << 6);
        const int xr = (r & 3) << 4;
#pragma unroll
        for (int s = 0; s < 4; ++s)
            *(uint4*)(dst + ((s * 16) ^ xr)) = tq[s];
    }
    if (tid < M_TILE) {
        for (int t = 0; t < K_TOP; ++t) { topv[tid * 65 + t] = -1.0f; topi[tid * 65 + t] = 0; }
    }
    __syncthreads();

    float vmin   = -1.0f;   // register-cached current min of this row's top-64
    int   minpos = 0;

    // Per-lane A-fragment LDS base pointers (row = t*16 + l16).
    // A 16x32 bf16 layout: lanes 0-15 take K {0..15}, lanes 16-31 take {16..31}.
    const char* a_base[2][2];
#pragma unroll
    for (int t = 0; t < 2; ++t) {
        const int r    = t * 16 + l16;
        const int xr   = (r & 3) << 4;
        const int rsel = hi16 << 5;
        a_base[t][0] = xtile + (r << 6) + (rsel ^ xr);
        a_base[t][1] = xtile + (r << 6) + (((16 + rsel)) ^ xr);
    }

    for (int n0 = 0; n0 < D_SAE; n0 += N_CHUNK) {
        // B column for this lane: 32x16 bf16 B-frag, lanes 0-15 hold K 0..15,
        // lanes 16-31 hold K 16..31 (W_enc rows are K-contiguous f32).
        const int    nl   = n0 + wave * 16 + l16;
        const float* wrow = W_enc + (size_t)nl * D_IN + hi16 * 16;

        v8f acc[2] = {};
        const char* a0l = a_base[0][0]; const char* a0h = a_base[0][1];
        const char* a1l = a_base[1][0]; const char* a1h = a_base[1][1];

#pragma unroll 2
        for (int kk = 0; kk < D_IN; kk += 32) {
            const float4* wq = (const float4*)(wrow + kk);
            const float4 f0 = wq[0], f1 = wq[1], f2 = wq[2], f3 = wq[3];
            __builtin_prefetch(wrow + kk + 512, 0, 1);   // global_prefetch_b8, 16 k-steps ahead

            v16bf bf;
            bf[0]=(__bf16)f0.x;  bf[1]=(__bf16)f0.y;  bf[2]=(__bf16)f0.z;  bf[3]=(__bf16)f0.w;
            bf[4]=(__bf16)f1.x;  bf[5]=(__bf16)f1.y;  bf[6]=(__bf16)f1.z;  bf[7]=(__bf16)f1.w;
            bf[8]=(__bf16)f2.x;  bf[9]=(__bf16)f2.y;  bf[10]=(__bf16)f2.z; bf[11]=(__bf16)f2.w;
            bf[12]=(__bf16)f3.x; bf[13]=(__bf16)f3.y; bf[14]=(__bf16)f3.z; bf[15]=(__bf16)f3.w;

            union { uint4 q[2]; v16bf v; } ua;
            ua.q[0] = *(const uint4*)a0l; ua.q[1] = *(const uint4*)a0h;
            acc[0] = __builtin_amdgcn_wmma_f32_16x16x32_bf16(
                         false, ua.v, false, bf, (short)0, acc[0], false, false);
            ua.q[0] = *(const uint4*)a1l; ua.q[1] = *(const uint4*)a1h;
            acc[1] = __builtin_amdgcn_wmma_f32_16x16x32_bf16(
                         false, ua.v, false, bf, (short)0, acc[1], false, false);

            a0l += M_TILE * 64; a0h += M_TILE * 64;      // next k-chunk (+2048B)
            a1l += M_TILE * 64; a1h += M_TILE * 64;
        }

        // Stage 32 rows x 128 latents of pre-activations (C layout: VGPR i ->
        // row base+i (lanes 0-15) / base+8+i (lanes 16-31), col = lane&15).
        const int cl = wave * 16 + l16;
#pragma unroll
        for (int t = 0; t < 2; ++t) {
            const int rb = t * 16 + (hi16 << 3);
#pragma unroll
            for (int i = 0; i < 8; ++i)
                stage[(rb + i) * 129 + cl] = acc[t][i];
        }
        __syncthreads();

        // One thread per row: bias + relu + running top-64 insert.
        if (tid < M_TILE) {
            float*          tv   = topv + tid * 65;
            unsigned short* ti   = topi + tid * 65;
            const float*    srow = stage + tid * 129;
            for (int j = 0; j < N_CHUNK; ++j) {
                float v = srow[j] + b_enc[n0 + j];
                v = fmaxf(v, 0.0f);
                if (v > vmin) {
                    tv[minpos] = v;
                    ti[minpos] = (unsigned short)(n0 + j);
                    float m = tv[0]; int mp = 0;
                    for (int t = 1; t < K_TOP; ++t) {
                        const float q = tv[t];
                        if (q < m) { m = q; mp = t; }
                    }
                    vmin = m; minpos = mp;
                }
            }
        }
        __syncthreads();
    }

    if (tid < M_TILE) {
        float* ov = out_vals + (size_t)(row0 + tid) * K_TOP;
        int*   oi = out_idx  + (size_t)(row0 + tid) * K_TOP;
        for (int t = 0; t < K_TOP; ++t) {
            ov[t] = topv[tid * 65 + t];
            oi[t] = (int)topi[tid * 65 + t];
        }
    }
}

// ============================================================================
// Kernel 2: decode — out[b,:] = b_dec + sum_k vals[b,k] * W_dec[idx[b,k],:]
// ============================================================================
__global__ __launch_bounds__(256)
void sae_decode(const float* __restrict__ vals,
                const int*   __restrict__ idx,
                const float* __restrict__ W_dec,
                const float* __restrict__ b_dec,
                float* __restrict__ out)
{
    const int b   = blockIdx.x;
    const int tid = threadIdx.x;
    const int c0  = tid * 4;                 // cols [c0,c0+4) and [1024+c0,..)

    float4 a0 = *(const float4*)(b_dec + c0);
    float4 a1 = *(const float4*)(b_dec + 1024 + c0);

    const float* vr = vals + (size_t)b * K_TOP;
    const int*   ir = idx  + (size_t)b * K_TOP;

#pragma unroll 4
    for (int k = 0; k < K_TOP; ++k) {
        const float  v = vr[k];              // block-uniform -> scalar loads
        const float* w = W_dec + (size_t)ir[k] * D_IN;
        const float4 w0 = *(const float4*)(w + c0);
        const float4 w1 = *(const float4*)(w + 1024 + c0);
        a0.x = fmaf(v, w0.x, a0.x); a0.y = fmaf(v, w0.y, a0.y);
        a0.z = fmaf(v, w0.z, a0.z); a0.w = fmaf(v, w0.w, a0.w);
        a1.x = fmaf(v, w1.x, a1.x); a1.y = fmaf(v, w1.y, a1.y);
        a1.z = fmaf(v, w1.z, a1.z); a1.w = fmaf(v, w1.w, a1.w);
    }

    float* o = out + (size_t)b * D_IN;
    *(float4*)(o + c0)        = a0;
    *(float4*)(o + 1024 + c0) = a1;
}

// ============================================================================
extern "C" void kernel_launch(void* const* d_in, const int* in_sizes, int n_in,
                              void* d_out, int out_size, void* d_ws, size_t ws_size,
                              hipStream_t stream)
{
    const float* x     = (const float*)d_in[0];
    const float* W_enc = (const float*)d_in[1];
    const float* b_enc = (const float*)d_in[2];
    const float* W_dec = (const float*)d_in[3];
    const float* b_dec = (const float*)d_in[4];
    // d_in[5] = k (== 64, compile-time K_TOP)

    float* vals = (float*)d_ws;                                          // 1 MB
    int*   idx  = (int*)((char*)d_ws + (size_t)B_TOK * K_TOP * 4);       // 1 MB

    (void)hipFuncSetAttribute(reinterpret_cast<const void*>(&sae_encode_topk),
                              hipFuncAttributeMaxDynamicSharedMemorySize, LDS_TOTAL);

    sae_encode_topk<<<B_TOK / M_TILE, 256, LDS_TOTAL, stream>>>(
        x, W_enc, b_enc, b_dec, vals, idx);
    sae_decode<<<B_TOK, 256, 0, stream>>>(vals, idx, W_dec, b_dec, (float*)d_out);
}
